// LSTM_67989332296302
// MI455X (gfx1250) — matrix operand order repaired and checked by reading, not compiled, not run on hardware
//
#include <hip/hip_runtime.h>
#include <hip/hip_bf16.h>
#include <math.h>

// Problem constants (must match the reference)
#define VV     100000
#define E      300
#define EPAD   320          // K padded to 10 x 32
#define EWORDS 160          // EPAD/2 packed bf16 pairs per row
#define H      768
#define FOURH  3072
#define TT     20
#define SS     8192

#define NWG    192          // persistent workgroups in the recurrence
#define HID_PER_WG 8        // 192 * 8 = 1536 = 2 dirs * 768 hidden units
#define ROWS_PER_WG 32      // 4 gates * 8 hidden units
#define HW     (H / 2)      // packed bf16 words per W_hh row

typedef __bf16 bf16_t;
typedef __attribute__((ext_vector_type(2)))  __bf16 bf16x2;
typedef __attribute__((ext_vector_type(16))) __bf16 v16bf;
typedef __attribute__((ext_vector_type(8)))  float  v8f;

__device__ __forceinline__ unsigned pkbf(float x, float y) {
    bf16x2 p;
    p.x = (__bf16)x;
    p.y = (__bf16)y;
    return __builtin_bit_cast(unsigned, p);
}
__device__ __forceinline__ float sigmoidf_(float x) { return 1.0f / (1.0f + __expf(-x)); }

// ---------------------------------------------------------------------------
// Prep 1: Xpad[s][0:160] = packed bf16 of emb[idx[s]][k], zero-padded to K=320.
// ---------------------------------------------------------------------------
__global__ void pack_x_kernel(const int* __restrict__ inp,
                              const float* __restrict__ emb,
                              unsigned* __restrict__ Xpad)
{
    const int i = blockIdx.x * 256 + threadIdx.x;     // over 8192*160
    if (i >= SS * EWORDS) return;
    const int row = i / EWORDS, w = i - row * EWORDS;
    const int k = 2 * w;
    const float* __restrict__ er = emb + (size_t)inp[row] * E;
    const float a0 = (k     < E) ? er[k]     : 0.0f;
    const float a1 = (k + 1 < E) ? er[k + 1] : 0.0f;
    Xpad[i] = pkbf(a0, a1);
}

// Prep 2: Wpad[n][0:160] = packed bf16 of W_ih[n][k], zero-padded to K=320.
__global__ void pack_w_kernel(const float* __restrict__ W_ih,
                              unsigned* __restrict__ Wpad)
{
    const int i = blockIdx.x * 256 + threadIdx.x;     // over 3072*160
    if (i >= FOURH * EWORDS) return;
    const int row = i / EWORDS, w = i - row * EWORDS;
    const int k = 2 * w;
    const float* __restrict__ wr = W_ih + (size_t)row * E;
    const float a0 = (k     < E) ? wr[k]     : 0.0f;
    const float a1 = (k + 1 < E) ? wr[k + 1] : 0.0f;
    Wpad[i] = pkbf(a0, a1);
}

// ---------------------------------------------------------------------------
// Kernel 1: gates = Xpad @ Wpad^T + (b_ih + b_hh), bf16 WMMA, fp32 accumulate.
// Each wave: one 16(M) x 64(N) strip = 4 accumulators sharing the A fragment.
// Software-pipelined: iteration kk+1's 10 b128 fragment loads are issued
// BEFORE iteration kk's 4 WMMAs, so matrix ops always run with the next
// step's loads in flight.
// Fragment words (straight off b128 loads, contiguous k = 16*hi + s, same for A and B):
//   A pair-word(v): K0/2 + hi*8 + v                 -> two aligned uint4 runs
//   B pair-word(v): K0/2 + hi*8 + v                 -> two aligned uint4 runs
// grid = (12, 512), block = 128 (4 waves = 4 adjacent 64-col strips).
// ---------------------------------------------------------------------------
struct Frag { union { v16bf v; uint4 q[2]; }; };
struct Step { Frag a, b0, b1, b2, b3; };

__device__ __forceinline__ void load_step(Step& s, int kk, int hi,
                                          const uint4* __restrict__ xr,
                                          const uint4* __restrict__ wr0,
                                          const uint4* __restrict__ wr1,
                                          const uint4* __restrict__ wr2,
                                          const uint4* __restrict__ wr3)
{
    const int ai = kk * 4 + 2 * hi;
    const int bi = kk * 4 + 2 * hi;
    s.a.q[0]  = xr[ai];
    s.a.q[1]  = xr[ai + 1];
    s.b0.q[0] = wr0[bi];  s.b0.q[1] = wr0[bi + 1];
    s.b1.q[0] = wr1[bi];  s.b1.q[1] = wr1[bi + 1];
    s.b2.q[0] = wr2[bi];  s.b2.q[1] = wr2[bi + 1];
    s.b3.q[0] = wr3[bi];  s.b3.q[1] = wr3[bi + 1];
}

__global__ void gates_gemm_kernel(const unsigned* __restrict__ Xpad,   // (S, 160)
                                  const unsigned* __restrict__ Wpad,   // (4H, 160)
                                  const float* __restrict__ b_ih,
                                  const float* __restrict__ b_hh,
                                  float* __restrict__ gates)           // (S, 4H)
{
    const int lane = threadIdx.x & 31;
    const int wv   = threadIdx.x >> 5;
    const int l    = lane & 15;
    const int hi   = lane >> 4;
    const int tm   = blockIdx.y;                          // 0..511
    const int tn0  = (blockIdx.x * 4 + wv) * 4;           // first of 4 N tiles

    const uint4* __restrict__ xr  = (const uint4*)(Xpad + (size_t)(tm * 16 + l) * EWORDS);
    const uint4* __restrict__ wr0 = (const uint4*)(Wpad + (size_t)((tn0 + 0) * 16 + l) * EWORDS);
    const uint4* __restrict__ wr1 = (const uint4*)(Wpad + (size_t)((tn0 + 1) * 16 + l) * EWORDS);
    const uint4* __restrict__ wr2 = (const uint4*)(Wpad + (size_t)((tn0 + 2) * 16 + l) * EWORDS);
    const uint4* __restrict__ wr3 = (const uint4*)(Wpad + (size_t)((tn0 + 3) * 16 + l) * EWORDS);

    v8f acc0 = {}, acc1 = {}, acc2 = {}, acc3 = {};

    Step cur, nxt;
    load_step(cur, 0, hi, xr, wr0, wr1, wr2, wr3);
#pragma unroll
    for (int kk = 0; kk < 10; ++kk) {
        if (kk < 9) load_step(nxt, kk + 1, hi, xr, wr0, wr1, wr2, wr3);
        acc0 = __builtin_amdgcn_wmma_f32_16x16x32_bf16(false, cur.a.v, false, cur.b0.v,
                                                       (short)0, acc0, false, false);
        acc1 = __builtin_amdgcn_wmma_f32_16x16x32_bf16(false, cur.a.v, false, cur.b1.v,
                                                       (short)0, acc1, false, false);
        acc2 = __builtin_amdgcn_wmma_f32_16x16x32_bf16(false, cur.a.v, false, cur.b2.v,
                                                       (short)0, acc2, false, false);
        acc3 = __builtin_amdgcn_wmma_f32_16x16x32_bf16(false, cur.a.v, false, cur.b3.v,
                                                       (short)0, acc3, false, false);
        cur = nxt;   // fully unrolled: pure register renaming
    }

    const int mbase = tm * 16 + hi * 8;
#pragma unroll
    for (int c = 0; c < 4; ++c) {
        const v8f acc = (c == 0) ? acc0 : (c == 1) ? acc1 : (c == 2) ? acc2 : acc3;
        const int n = (tn0 + c) * 16 + l;
        const float bias = b_ih[n] + b_hh[n];
#pragma unroll
        for (int r = 0; r < 8; ++r)                   // C/D layout: M = r + hi*8, N = l
            gates[(size_t)(mbase + r) * FOURH + n] = acc[r] + bias;
    }
}

// ---------------------------------------------------------------------------
// Init: zero h ping-pong buffers and the grid barrier state.
// ---------------------------------------------------------------------------
__global__ void init_kernel(float* __restrict__ hbuf, unsigned* __restrict__ bar)
{
    const int i = blockIdx.x * blockDim.x + threadIdx.x;
    if (i < 4 * H) hbuf[i] = 0.0f;    // [2 dirs][2 parities][H]
    if (i < 2)     bar[i]  = 0u;      // [0]=arrive count (monotonic), [1]=release step
}

// ---------------------------------------------------------------------------
// Kernel 2: persistent bidirectional-LSTM recurrence.
// Each WG owns 8 hidden units of one direction: 32 rows of W_hh pinned in LDS
// as packed bf16 pairs (48 KB). One global barrier per timestep; h is
// double-buffered by parity. bf16 -> f32 expansion is a pure shift (bits<<16).
// ---------------------------------------------------------------------------
__global__ void __launch_bounds__(256, 1)
lstm_recurrence_kernel(const float* __restrict__ gates_f,
                       const float* __restrict__ gates_b,
                       const float* __restrict__ W_hh_f,   // (4H, H)
                       const float* __restrict__ W_hh_b,   // (4H, H)
                       float* __restrict__ hbuf,           // [2 dirs][2 parity][H]
                       unsigned* __restrict__ bar)
{
    __shared__ unsigned wlds[ROWS_PER_WG * HW];          // 48 KB packed bf16 slab
    __shared__ __align__(16) float hlds[H];              // staged h_{t-1}
    __shared__ float gdot[ROWS_PER_WG];
    __shared__ float cst[HID_PER_WG];

    const int tid    = threadIdx.x;
    const int wg     = blockIdx.x;             // 0..191
    const int d      = wg / 96;                // direction
    const int base_j = (wg % 96) * HID_PER_WG;
    const float* __restrict__ Whh   = d ? W_hh_b : W_hh_f;
    const float* __restrict__ gates = d ? gates_b : gates_f;
    float* __restrict__ hme = hbuf + d * (2 * H);

    // Stage W_hh rows {g*H + base_j + jl : g<4, jl<8} into LDS (one time).
    for (int e = tid; e < ROWS_PER_WG * HW; e += 256) {
        const int r = e / HW, w = e - r * HW;
        const int g = r >> 3, jl = r & 7;
        const float* __restrict__ src = Whh + (size_t)(g * H + base_j + jl) * H + 2 * w;
        wlds[e] = pkbf(src[0], src[1]);
    }
    if (tid < HID_PER_WG) cst[tid] = 0.0f;
    __syncthreads();

    const int lane = tid & 31, wv = tid >> 5;   // 8 waves, 4 rows each

    for (int t = 0; t < SS; ++t) {
        const int cur = t & 1, nxt = cur ^ 1;

        for (int k = tid; k < H; k += 256) hlds[k] = hme[cur * H + k];
        __syncthreads();

        // 32 recurrent dot products of length 768 (packed-pair FMAs).
#pragma unroll
        for (int i = 0; i < 4; ++i) {
            const int r = wv * 4 + i;
            const unsigned* __restrict__ wrow = wlds + r * HW;
            float s = 0.0f;
#pragma unroll 4
            for (int w = lane; w < HW; w += 32) {
                const unsigned u = wrow[w];
                const float w0 = __builtin_bit_cast(float, u << 16);
                const float w1 = __builtin_bit_cast(float, u & 0xffff0000u);
                const float2 hv = *(const float2*)(hlds + 2 * w);
                s += w0 * hv.x + w1 * hv.y;
            }
#pragma unroll
            for (int off = 16; off > 0; off >>= 1) s += __shfl_xor(s, off, 32);
            if (lane == 0) gdot[r] = s;
        }
        __syncthreads();

        // LSTM cell update for the 8 owned hidden units (gate order i,f,g,o).
        if (tid < HID_PER_WG) {
            const float* __restrict__ gb = gates + (size_t)t * FOURH + base_j + tid;
            const float gi = gb[0 * H] + gdot[0 * HID_PER_WG + tid];
            const float gf = gb[1 * H] + gdot[1 * HID_PER_WG + tid];
            const float gg = gb[2 * H] + gdot[2 * HID_PER_WG + tid];
            const float go = gb[3 * H] + gdot[3 * HID_PER_WG + tid];
            const float c  = sigmoidf_(gf) * cst[tid] + sigmoidf_(gi) * tanhf(gg);
            cst[tid] = c;
            hme[nxt * H + base_j + tid] = sigmoidf_(go) * tanhf(c);
            if (t + 1 < SS) __builtin_prefetch(gb + FOURH, 0, 1);  // global_prefetch_b8
        }
        __syncthreads();

        // Grid-wide barrier: monotonic arrive counter + release-step flag.
        if (tid == 0) {
            __threadfence();
            const unsigned prev =
                __hip_atomic_fetch_add(&bar[0], 1u, __ATOMIC_ACQ_REL, __HIP_MEMORY_SCOPE_AGENT);
            if (prev + 1u == (unsigned)NWG * (unsigned)(t + 1)) {
                __hip_atomic_store(&bar[1], (unsigned)(t + 1),
                                   __ATOMIC_RELEASE, __HIP_MEMORY_SCOPE_AGENT);
            } else {
                while (__hip_atomic_load(&bar[1], __ATOMIC_ACQUIRE,
                                         __HIP_MEMORY_SCOPE_AGENT) < (unsigned)(t + 1)) {
                    __builtin_amdgcn_s_sleep(2);
                }
            }
            __threadfence();
        }
        __syncthreads();
    }
}

// ---------------------------------------------------------------------------
// Kernel 3: out = concat(h_f, h_b) @ W_tag^T + b_tag  (20 x 1536, trivial).
// ---------------------------------------------------------------------------
__global__ void tag_kernel(const float* __restrict__ hbuf,
                           const float* __restrict__ W_tag,  // (T, 2H)
                           const float* __restrict__ b_tag,  // (T)
                           float* __restrict__ out)          // (T)
{
    const int r = threadIdx.x;
    if (r >= TT) return;
    const int fp = ((SS - 1) & 1) ^ 1;   // parity holding h_{S-1}
    const float* hf = hbuf + 0 * (2 * H) + fp * H;
    const float* hb = hbuf + 1 * (2 * H) + fp * H;
    const float* wr = W_tag + (size_t)r * (2 * H);
    float s = b_tag[r];
    for (int k = 0; k < H; ++k) s += hf[k] * wr[k];
    for (int k = 0; k < H; ++k) s += hb[k] * wr[H + k];
    out[r] = s;
}

// ---------------------------------------------------------------------------
extern "C" void kernel_launch(void* const* d_in, const int* in_sizes, int n_in,
                              void* d_out, int out_size, void* d_ws, size_t ws_size,
                              hipStream_t stream)
{
    const int*   inp    = (const int*)  d_in[0];
    const float* emb    = (const float*)d_in[1];
    const float* W_ih_f = (const float*)d_in[2];
    const float* W_hh_f = (const float*)d_in[3];
    const float* b_ih_f = (const float*)d_in[4];
    const float* b_hh_f = (const float*)d_in[5];
    const float* W_ih_b = (const float*)d_in[6];
    const float* W_hh_b = (const float*)d_in[7];
    const float* b_ih_b = (const float*)d_in[8];
    const float* b_hh_b = (const float*)d_in[9];
    const float* W_tag  = (const float*)d_in[10];
    const float* b_tag  = (const float*)d_in[11];
    float* out = (float*)d_out;

    // Workspace layout: gates_f | gates_b | hbuf | bar(pad) | Xpad | Wpad_f | Wpad_b
    char* ws = (char*)d_ws;
    size_t off = 0;
    float* gates_f = (float*)(ws + off);  off += (size_t)SS * FOURH * sizeof(float);
    float* gates_b = (float*)(ws + off);  off += (size_t)SS * FOURH * sizeof(float);
    float* hbuf    = (float*)(ws + off);  off += (size_t)4 * H * sizeof(float);
    unsigned* bar  = (unsigned*)(ws + off); off += 256;  // 2 words + pad, keeps 16B align
    unsigned* Xpad   = (unsigned*)(ws + off); off += (size_t)SS * EWORDS * sizeof(unsigned);
    unsigned* Wpad_f = (unsigned*)(ws + off); off += (size_t)FOURH * EWORDS * sizeof(unsigned);
    unsigned* Wpad_b = (unsigned*)(ws + off); off += (size_t)FOURH * EWORDS * sizeof(unsigned);

    init_kernel<<<(4 * H + 255) / 256, 256, 0, stream>>>(hbuf, bar);

    pack_x_kernel<<<(SS * EWORDS + 255) / 256, 256, 0, stream>>>(inp, emb, Xpad);
    pack_w_kernel<<<(FOURH * EWORDS + 255) / 256, 256, 0, stream>>>(W_ih_f, Wpad_f);
    pack_w_kernel<<<(FOURH * EWORDS + 255) / 256, 256, 0, stream>>>(W_ih_b, Wpad_b);

    dim3 g(12, 512, 1);  // 48 wave-strips of 64 cols (4/block) x 512 M tiles
    gates_gemm_kernel<<<g, 128, 0, stream>>>(Xpad, Wpad_f, b_ih_f, b_hh_f, gates_f);
    gates_gemm_kernel<<<g, 128, 0, stream>>>(Xpad, Wpad_b, b_ih_b, b_hh_b, gates_b);

    lstm_recurrence_kernel<<<NWG, 256, 0, stream>>>(gates_f, gates_b,
                                                    W_hh_f, W_hh_b, hbuf, bar);

    tag_kernel<<<1, 32, 0, stream>>>(hbuf, W_tag, b_tag, out);
}
